// SparseMoE_55190329753810
// MI455X (gfx1250) — matrix operand order repaired and checked by reading, not compiled, not run on hardware
//
#include <hip/hip_runtime.h>

typedef _Float16 half8  __attribute__((ext_vector_type(8)));
typedef _Float16 half16 __attribute__((ext_vector_type(16)));
typedef float    v8f    __attribute__((ext_vector_type(8)));

#define N_TOK   65536
#define DIM     512
#define NEXP    10
#define TM      128
#define PADTOT  (N_TOK + NEXP * TM)   // 66816 padded slots
#define NTILES  (PADTOT / TM)         // 522 tiles
#define KTILES  (DIM / 32)            // 16

// meta (int) layout: [0..9]=counts  [16..26]=po (padded offsets)  [32..41]=cursor

// ---------------------------------------------------------------- routing ----
__global__ void moe_init_meta(int* __restrict__ meta, int* __restrict__ perm) {
    int t = blockIdx.x * blockDim.x + threadIdx.x;
    if (t < 64) meta[t] = 0;
    if (t < PADTOT) perm[t] = -1;
}

__global__ void moe_route(const float* __restrict__ mw,
                          int* __restrict__ eidx, float* __restrict__ gate,
                          int* __restrict__ counts) {
    int t = blockIdx.x * blockDim.x + threadIdx.x;
    if (t >= N_TOK) return;
    const float* p = mw + (size_t)t * NEXP;
    float m = p[0]; int best = 0;
#pragma unroll
    for (int i = 1; i < NEXP; ++i) {
        float v = p[i];
        if (v > m) { m = v; best = i; }   // strict '>' == first-max (jnp.argmax)
    }
    eidx[t] = best;
    gate[t] = m;
    atomicAdd(&counts[best], 1);
}

__global__ void moe_scan(int* __restrict__ meta) {
    if (threadIdx.x == 0 && blockIdx.x == 0) {
        int off = 0;
        for (int e = 0; e < NEXP; ++e) {
            meta[16 + e] = off;           // po[e]
            meta[32 + e] = off;           // cursor[e]
            off += ((meta[e] + TM - 1) / TM) * TM;
        }
        meta[16 + NEXP] = off;            // po[E] = padded total
    }
}

__global__ void moe_fill(const int* __restrict__ eidx, const float* __restrict__ gate,
                         int* __restrict__ cursor, int* __restrict__ perm,
                         float* __restrict__ pgate) {
    int t = blockIdx.x * blockDim.x + threadIdx.x;
    if (t >= N_TOK) return;
    int e = eidx[t];
    int s = atomicAdd(&cursor[e], 1);
    perm[s]  = t;
    pgate[s] = gate[t];
}

// ------------------------------------------- weight transpose + f16 convert --
// Wt[e][n][k] = (f16) W[e][k][n]   (so B-fragments read contiguous K per column)
__global__ void moe_prep_w(const float* __restrict__ W, _Float16* __restrict__ Wt) {
    int idx = blockIdx.x * blockDim.x + threadIdx.x;       // E*512*128 threads
    if (idx >= NEXP * DIM * (DIM / 4)) return;
    int k0 = (idx & 127) * 4;
    int n  = (idx >> 7) & (DIM - 1);
    int e  = idx >> 16;                                    // / (128*512)
    const float* src = W + (size_t)e * DIM * DIM + n;
    _Float16*    dst = Wt + (size_t)e * DIM * DIM + (size_t)n * DIM + k0;
#pragma unroll
    for (int q = 0; q < 4; ++q)
        dst[q] = (_Float16)src[(size_t)(k0 + q) * DIM];
}

// ------------------------------------------------------------- fused GEMMs ---
struct SMem {
    _Float16 A[TM][DIM + 8];      // gathered x tile  (f16)   133,120 B
    _Float16 H[TM][DIM + 8];      // hidden tile      (f16)   133,120 B
    _Float16 B[2][128][40];       // double-buffered W tile    20,480 B
    int      perm[TM];
    float    gate[TM];
};

__device__ __forceinline__ v8f wmma16(half16 a, half16 b, v8f c) {
    return __builtin_amdgcn_wmma_f32_16x16x32_f16(false, a, false, b,
                                                  (short)0, c, false, false);
}

// async global->LDS copy, 16 bytes per lane, tracked by ASYNCcnt
__device__ __forceinline__ void async_lds_b128(unsigned lds_addr, unsigned voff,
                                               const _Float16* sbase) {
    asm volatile("global_load_async_to_lds_b128 %0, %1, %2"
                 :: "v"(lds_addr), "v"(voff), "s"(sbase)
                 : "memory");
}
__device__ __forceinline__ void wait_async_le2() {
    asm volatile("s_wait_asynccnt 0x2" ::: "memory");
}
__device__ __forceinline__ void wait_async_0() {
    asm volatile("s_wait_asynccnt 0x0" ::: "memory");
}

// issue one 128x32 (f16) B tile: 512 x 16B chunks, 2 per thread
__device__ __forceinline__ void issue_btile(const _Float16* wt, unsigned base_off,
                                            int kt, _Float16 (*Bb)[40], int tid) {
#pragma unroll
    for (int i = 0; i < 2; ++i) {
        int idx = tid + i * 256;
        int n = idx >> 2, c8 = idx & 3;
        unsigned lds  = (unsigned)(uintptr_t)&Bb[n][c8 * 8];
        unsigned voff = base_off + (unsigned)((n * DIM + kt * 32 + c8 * 8) * 2);
        async_lds_b128(lds, voff, wt);
    }
}

// one 128x128 output chunk: 16 k-tiles with async double-buffered B stream
__device__ __forceinline__ void run_gemm(
    const _Float16 (*Asrc)[DIM + 8], _Float16 (*Bb)[128][40],
    const _Float16* wt, unsigned base_off,
    int tid, int mrow, int ncol, int lhalf, int l15,
    v8f (&acc)[2][4])
{
    issue_btile(wt, base_off, 0, Bb[0], tid);
    for (int kt = 0; kt < KTILES; ++kt) {
        if (kt + 1 < KTILES) {
            issue_btile(wt, base_off, kt + 1, Bb[(kt + 1) & 1], tid);
            wait_async_le2();              // oldest 2 chunks (tile kt) landed
        } else {
            wait_async_0();
        }
        __syncthreads();                   // all waves' tile-kt chunks in LDS

        half16 af[2], bf[4];
        const _Float16(*Bc)[40] = Bb[kt & 1];
#pragma unroll
        for (int mi = 0; mi < 2; ++mi) {
            int M  = mrow + mi * 16 + l15;
            int kb = kt * 32 + lhalf * 8;
            *(half8*)&af[mi]       = *(const half8*)&Asrc[M][kb];
            *((half8*)&af[mi] + 1) = *(const half8*)&Asrc[M][kb + 16];
        }
#pragma unroll
        for (int ni = 0; ni < 4; ++ni) {
            int Nc = ncol + ni * 16 + l15;
            int kb = lhalf * 8;
            *(half8*)&bf[ni]       = *(const half8*)&Bc[Nc][kb];
            *((half8*)&bf[ni] + 1) = *(const half8*)&Bc[Nc][kb + 16];
        }
#pragma unroll
        for (int mi = 0; mi < 2; ++mi)
#pragma unroll
            for (int ni = 0; ni < 4; ++ni)
                acc[mi][ni] = wmma16(af[mi], bf[ni], acc[mi][ni]);

        __syncthreads();                   // done reading before buf reuse
    }
}

__global__ __launch_bounds__(256) void moe_fused_gemm(
    const float* __restrict__ x,
    const float* __restrict__ b1, const float* __restrict__ b2,
    const _Float16* __restrict__ wt1, const _Float16* __restrict__ wt2,
    const int* __restrict__ perm, const float* __restrict__ pgate,
    const int* __restrict__ meta, float* __restrict__ out)
{
    __shared__ SMem sm;
    const int tid  = threadIdx.x;
    const int row0 = blockIdx.x * TM;

    // which expert segment does this tile belong to?
    int e = -1;
#pragma unroll
    for (int i = 0; i < NEXP; ++i) {
        int s = meta[16 + i], t = meta[16 + i + 1];
        if (row0 >= s && row0 < t) e = i;
    }
    if (e < 0) return;

    if (tid < TM) {
        sm.perm[tid] = perm[row0 + tid];
        sm.gate[tid] = pgate[row0 + tid];
    }
    __syncthreads();

    // gather + convert x rows -> sm.A (zero rows for padding slots)
    for (int i = tid; i < TM * (DIM / 8); i += 256) {
        int r  = i >> 6;
        int c8 = i & 63;
        int tok = sm.perm[r];
        half8 h;
        if (tok >= 0) {
            const float4* src = (const float4*)(x + (size_t)tok * DIM) + c8 * 2;
            float4 f0 = src[0], f1 = src[1];
            h[0]=(_Float16)f0.x; h[1]=(_Float16)f0.y; h[2]=(_Float16)f0.z; h[3]=(_Float16)f0.w;
            h[4]=(_Float16)f1.x; h[5]=(_Float16)f1.y; h[6]=(_Float16)f1.z; h[7]=(_Float16)f1.w;
        } else {
#pragma unroll
            for (int q = 0; q < 8; ++q) h[q] = (_Float16)0.0f;
        }
        *(half8*)&sm.A[r][c8 * 8] = h;
    }

    const int wave  = tid >> 5;           // wave32: 8 waves
    const int lane  = tid & 31;
    const int mrow  = (wave & 3) * 32;    // 4x2 wave grid over 128x128 chunk
    const int ncol  = (wave >> 2) * 64;
    const int lhalf = lane >> 4;          // 0|1
    const int l15   = lane & 15;

    __syncthreads();                      // sm.A ready

    // ===================== GEMM1:  H = A @ W1 + b1 =====================
    for (int nc = 0; nc < 4; ++nc) {
        v8f acc[2][4];
#pragma unroll
        for (int mi = 0; mi < 2; ++mi)
#pragma unroll
            for (int ni = 0; ni < 4; ++ni)
#pragma unroll
                for (int q = 0; q < 8; ++q) acc[mi][ni][q] = 0.0f;

        unsigned base_off = (unsigned)(((e * DIM + nc * 128) * DIM) * 2);
        run_gemm(sm.A, sm.B, wt1, base_off, tid, mrow, ncol, lhalf, l15, acc);

        // epilogue: h = acc + b1  -> sm.H (f16)
#pragma unroll
        for (int mi = 0; mi < 2; ++mi)
#pragma unroll
            for (int ni = 0; ni < 4; ++ni) {
                int ng = nc * 128 + ncol + ni * 16 + l15;
                float bias = b1[(size_t)e * DIM + ng];
#pragma unroll
                for (int r = 0; r < 8; ++r) {
                    int M = mrow + mi * 16 + r + lhalf * 8;
                    sm.H[M][ng] = (_Float16)(acc[mi][ni][r] + bias);
                }
            }
    }
    __syncthreads();                      // sm.H complete

    // ============== GEMM2:  out = gate * (H @ W2 + b2), scattered ==============
    for (int nc = 0; nc < 4; ++nc) {
        v8f acc[2][4];
#pragma unroll
        for (int mi = 0; mi < 2; ++mi)
#pragma unroll
            for (int ni = 0; ni < 4; ++ni)
#pragma unroll
                for (int q = 0; q < 8; ++q) acc[mi][ni][q] = 0.0f;

        unsigned base_off = (unsigned)(((e * DIM + nc * 128) * DIM) * 2);
        run_gemm(sm.H, sm.B, wt2, base_off, tid, mrow, ncol, lhalf, l15, acc);

        // epilogue: scatter gate * (acc + b2) to out[token]
#pragma unroll
        for (int mi = 0; mi < 2; ++mi)
#pragma unroll
            for (int ni = 0; ni < 4; ++ni) {
                int ng = nc * 128 + ncol + ni * 16 + l15;
                float bias = b2[(size_t)e * DIM + ng];
#pragma unroll
                for (int r = 0; r < 8; ++r) {
                    int M = mrow + mi * 16 + r + lhalf * 8;
                    int tok = sm.perm[M];
                    if (tok >= 0)
                        out[(size_t)tok * DIM + ng] = sm.gate[M] * (acc[mi][ni][r] + bias);
                }
            }
    }
}

// ---------------------------------------------------------------- launcher ---
extern "C" void kernel_launch(void* const* d_in, const int* in_sizes, int n_in,
                              void* d_out, int out_size, void* d_ws, size_t ws_size,
                              hipStream_t stream) {
    (void)in_sizes; (void)n_in; (void)out_size; (void)ws_size;
    const float* x   = (const float*)d_in[0];
    const float* mw  = (const float*)d_in[1];
    const float* W1  = (const float*)d_in[2];
    const float* b1  = (const float*)d_in[3];
    const float* W2  = (const float*)d_in[4];
    const float* b2  = (const float*)d_in[5];
    float* out = (float*)d_out;

    char* ws = (char*)d_ws;
    const size_t WT_BYTES = (size_t)NEXP * DIM * DIM * sizeof(_Float16);  // 5,242,880
    _Float16* wt1  = (_Float16*)(ws);
    _Float16* wt2  = (_Float16*)(ws + WT_BYTES);
    int*      perm = (int*)   (ws + 2 * WT_BYTES);
    float*    pgat = (float*) (ws + 2 * WT_BYTES + (size_t)PADTOT * 4);
    int*      eidx = (int*)   (ws + 2 * WT_BYTES + (size_t)PADTOT * 8);
    float*    gate = (float*) (ws + 2 * WT_BYTES + (size_t)PADTOT * 8 + (size_t)N_TOK * 4);
    int*      meta = (int*)   (ws + 2 * WT_BYTES + (size_t)PADTOT * 8 + (size_t)N_TOK * 8);

    moe_init_meta<<<(PADTOT + 255) / 256, 256, 0, stream>>>(meta, perm);
    moe_route<<<(N_TOK + 255) / 256, 256, 0, stream>>>(mw, eidx, gate, meta);
    moe_scan<<<1, 1, 0, stream>>>(meta);
    moe_fill<<<(N_TOK + 255) / 256, 256, 0, stream>>>(eidx, gate, meta + 32, perm, pgat);

    const int prep_threads = NEXP * DIM * (DIM / 4);
    moe_prep_w<<<(prep_threads + 255) / 256, 256, 0, stream>>>(W1, wt1);
    moe_prep_w<<<(prep_threads + 255) / 256, 256, 0, stream>>>(W2, wt2);

    moe_fused_gemm<<<NTILES, 256, 0, stream>>>(x, b1, b2, wt1, wt2,
                                               perm, pgat, meta, out);
}
